// GRU_50440095924902
// MI455X (gfx1250) — matrix operand order repaired and checked by reading, not compiled, not run on hardware
//
#include <hip/hip_runtime.h>

#define T_STEPS 512
#define NENV    256
#define STATE   64
#define HID     128
#define G3      (3*HID)      // 384 gate columns (r,z,n)
#define EPB     16           // envs per block = one WMMA M-tile
#define NBLK    (NENV/EPB)   // 16 blocks (recurrence limits parallelism to envs)
#define THREADS 256          // 8 wave32 waves; wave w owns hidden cols [16w,16w+16)

typedef __attribute__((ext_vector_type(16))) __bf16       v16bf;
typedef __attribute__((ext_vector_type(8)))  float        v8f;
typedef __attribute__((ext_vector_type(4)))  float        v4f;
typedef __attribute__((ext_vector_type(4)))  unsigned int v4u;

union ABFrag { v4u u[2]; v16bf v; };

// A fragment: 16x32 bf16 tile (rows = envs) from row-major LDS matrix.
// Layout used here (contiguous K, same as load_B): lane = hi*16 + row; elem i -> K = k0 + 16*hi + i (16 contiguous K per lane)
__device__ __forceinline__ v16bf load_A(const __bf16* base, int stride, int k0, int lane) {
  const int row = lane & 15;
  const int hi  = lane >> 4;
  const __bf16* p = base + row * stride + k0 + hi * 16;
  ABFrag t;
  t.u[0] = *(const v4u*)(p);
  t.u[1] = *(const v4u*)(p + 8);
  return t.v;
}

// B fragment: 32x16 bf16 tile, B[k][n] = W[c0+n][k0+k] with W row-major (row = gate col).
// Layout used here (contiguous K, same as load_A): lane = hi*16 + n; elem i -> K = k0 + 16*hi + i (16 contiguous K per lane)
__device__ __forceinline__ v16bf load_B(const __bf16* W, int stride, int c0, int k0, int lane) {
  const int n  = lane & 15;
  const int hi = lane >> 4;
  const __bf16* p = W + (c0 + n) * stride + k0 + hi * 16;
  ABFrag t;
  t.u[0] = *(const v4u*)(p);
  t.u[1] = *(const v4u*)(p + 8);
  return t.v;
}

__device__ __forceinline__ v8f wmma_bf16(v16bf a, v16bf b, v8f c) {
  // D = A(16x32 bf16) x B(32x16 bf16) + C(16x16 f32)
  return __builtin_amdgcn_wmma_f32_16x16x32_bf16(false, a, false, b, (short)0, c, false, false);
}

// fast sigmoid: one v_exp_f32 + one v_rcp_f32 (avoid IEEE div_scale/NR chain)
__device__ __forceinline__ float sigmoid_f(float x) {
  return __builtin_amdgcn_rcpf(1.0f + __expf(-x));
}

// fast tanh: native V_TANH_F32 on gfx1250 if the builtin exists, else exp+rcp.
__device__ __forceinline__ float tanh_f(float x) {
#if defined(__has_builtin)
#if __has_builtin(__builtin_amdgcn_tanhf)
  return __builtin_amdgcn_tanhf(x);
#else
  const float e = __expf(2.0f * x);
  return 1.0f - 2.0f * __builtin_amdgcn_rcpf(e + 1.0f);
#endif
#else
  const float e = __expf(2.0f * x);
  return 1.0f - 2.0f * __builtin_amdgcn_rcpf(e + 1.0f);
#endif
}

__global__ __launch_bounds__(THREADS, 1)
void gru_scan_kernel(const float* __restrict__ x,
                     const float* __restrict__ hxs,
                     const float* __restrict__ masks,
                     const float* __restrict__ wih,
                     const float* __restrict__ whh,
                     const float* __restrict__ bih,
                     const float* __restrict__ bhh,
                     float* __restrict__ out)
{
  // LDS budget: 48K + 96K + 2K + 4K + 8K + 64B = ~158 KB (WGP holds 320 KB)
  __shared__ __align__(16) __bf16 sWih[G3 * STATE]; // W_ih, bf16, row-major (row = gate col)
  __shared__ __align__(16) __bf16 sWhh[G3 * HID];   // W_hh, bf16, row-major
  __shared__ __align__(16) __bf16 sX  [EPB * STATE];// x_t slice, bf16 (A operand)
  __shared__ __align__(16) __bf16 sHm [EPB * HID];  // masked h, bf16 (A operand)
  __shared__ __align__(16) float  sH  [EPB * HID];  // h, f32 (carried state)
  __shared__           float  sM  [EPB];            // per-env reset mask

  const int tid  = threadIdx.x;
  const int lane = tid & 31;
  const int wid  = tid >> 5;               // wave id 0..7
  const int e0   = blockIdx.x * EPB;       // first env of this block

  // ---- one-time: stage weights (f32 -> bf16) and initial hidden state ----
  for (int i = tid; i < (G3 * STATE) / 4; i += THREADS) {
    v4f v = ((const v4f*)wih)[i];
    __bf16* d = sWih + i * 4;
    d[0] = (__bf16)v.x; d[1] = (__bf16)v.y; d[2] = (__bf16)v.z; d[3] = (__bf16)v.w;
  }
  for (int i = tid; i < (G3 * HID) / 4; i += THREADS) {
    v4f v = ((const v4f*)whh)[i];
    __bf16* d = sWhh + i * 4;
    d[0] = (__bf16)v.x; d[1] = (__bf16)v.y; d[2] = (__bf16)v.z; d[3] = (__bf16)v.w;
  }
  for (int i = tid; i < (EPB * HID) / 4; i += THREADS) {
    ((v4f*)sH)[i] = ((const v4f*)(hxs + (size_t)e0 * HID))[i];
  }

  // per-lane constants: this lane owns gate column jcol for envs (hi*8 .. hi*8+7)
  const int jcol = wid * 16 + (lane & 15);        // hidden index
  const int cr = wid * 16;                        // r-gate tile base col
  const int cz = HID + wid * 16;                  // z-gate tile base col
  const int cn = 2 * HID + wid * 16;              // n-gate tile base col
  const float bxr = bih[jcol],           bhrv = bhh[jcol];
  const float bxz = bih[HID + jcol],     bhzv = bhh[HID + jcol];
  const float bxn = bih[2 * HID + jcol], bhnv = bhh[2 * HID + jcol];
  const int ebase = (lane >> 4) * 8;              // env base for C-tile elements

  // software pipeline: x/mask for step t live in registers one iteration early
  v4f  xv = ((const v4f*)(x + (size_t)e0 * STATE))[tid];
  float mv = (tid < EPB) ? masks[e0 + tid] : 0.0f;

  for (int t = 0; t < T_STEPS; ++t) {
    // 1. stage x_t (bf16) and mask from the pipelined registers
    {
      __bf16* d = sX + tid * 4;
      d[0] = (__bf16)xv.x; d[1] = (__bf16)xv.y; d[2] = (__bf16)xv.z; d[3] = (__bf16)xv.w;
      if (tid < EPB) sM[tid] = mv;
    }
    __syncthreads(); // sX/sM ready; prior ys store (reads sH) done before masking

    // 1b. kick off next step's global reads; latency hides under WMMA+pointwise
    if (t + 1 < T_STEPS) {
      xv = ((const v4f*)(x + (size_t)((t + 1) * NENV + e0) * STATE))[tid];
      if (tid < EPB) mv = masks[(t + 1) * NENV + e0 + tid];
    }

    // 2. inline episode reset: h *= m; produce bf16 A operand
    {
      const int e  = tid >> 4;
      const int jb = (tid & 15) * 8;
      const float m = sM[e];
#pragma unroll
      for (int q = 0; q < 8; ++q) {
        float hv = sH[e * HID + jb + q] * m;
        sH[e * HID + jb + q] = hv;
        sHm[e * HID + jb + q] = (__bf16)hv;
      }
    }
    __syncthreads(); // masked h (f32 + bf16) visible to all waves

    // 3. WMMA: gx = x@Wih^T (K=64), gh = h@Whh^T (K=128), kept separate for r*hn
    v8f axr = {0.f,0.f,0.f,0.f,0.f,0.f,0.f,0.f};
    v8f axz = axr, axn = axr, ahr = axr, ahz = axr, ahn = axr;
#pragma unroll
    for (int ks = 0; ks < STATE / 32; ++ks) {
      v16bf A = load_A(sX, STATE, ks * 32, lane);
      axr = wmma_bf16(A, load_B(sWih, STATE, cr, ks * 32, lane), axr);
      axz = wmma_bf16(A, load_B(sWih, STATE, cz, ks * 32, lane), axz);
      axn = wmma_bf16(A, load_B(sWih, STATE, cn, ks * 32, lane), axn);
    }
#pragma unroll
    for (int ks = 0; ks < HID / 32; ++ks) {
      v16bf A = load_A(sHm, HID, ks * 32, lane);
      ahr = wmma_bf16(A, load_B(sWhh, HID, cr, ks * 32, lane), ahr);
      ahz = wmma_bf16(A, load_B(sWhh, HID, cz, ks * 32, lane), ahz);
      ahn = wmma_bf16(A, load_B(sWhh, HID, cn, ks * 32, lane), ahn);
    }

    // 4. fused GRU pointwise update, entirely per-lane on C-tile layout:
    //    element i of each accumulator = (env ebase+i, hidden col jcol)
#pragma unroll
    for (int i = 0; i < 8; ++i) {
      const int e = ebase + i;
      const float r    = sigmoid_f(axr[i] + ahr[i] + bxr + bhrv);
      const float z    = sigmoid_f(axz[i] + ahz[i] + bxz + bhzv);
      const float cand = tanh_f(axn[i] + bxn + r * (ahn[i] + bhnv));
      const float hm   = sH[e * HID + jcol];           // already masked
      sH[e * HID + jcol] = (1.0f - z) * cand + z * hm; // torch GRU update
    }
    __syncthreads(); // full h_t assembled in LDS

    // 5. stream ys[t] out coalesced (2 x b128 per thread)
    {
      v4f* dst = (v4f*)(out + (size_t)(t * NENV + e0) * HID);
      const v4f* src = (const v4f*)sH;
      dst[tid]           = src[tid];
      dst[tid + THREADS] = src[tid + THREADS];
    }
  }

  // h_final, appended after ys region
  {
    v4f* dst = (v4f*)(out + (size_t)T_STEPS * NENV * HID + (size_t)e0 * HID);
    const v4f* src = (const v4f*)sH;
    dst[tid]           = src[tid];
    dst[tid + THREADS] = src[tid + THREADS];
  }
}

extern "C" void kernel_launch(void* const* d_in, const int* in_sizes, int n_in,
                              void* d_out, int out_size, void* d_ws, size_t ws_size,
                              hipStream_t stream) {
  (void)in_sizes; (void)n_in; (void)out_size; (void)d_ws; (void)ws_size;
  const float* x     = (const float*)d_in[0];
  const float* hxs   = (const float*)d_in[1];
  const float* masks = (const float*)d_in[2];
  const float* wih   = (const float*)d_in[3];
  const float* whh   = (const float*)d_in[4];
  const float* bih   = (const float*)d_in[5];
  const float* bhh   = (const float*)d_in[6];
  float* out = (float*)d_out;
  gru_scan_kernel<<<NBLK, THREADS, 0, stream>>>(x, hxs, masks, wih, whh, bih, bhh, out);
}
